// MultiHeadSelfAttention_26852135534953
// MI455X (gfx1250) — matrix operand order repaired and checked by reading, not compiled, not run on hardware
//
#include <hip/hip_runtime.h>
#include <hip/hip_bf16.h>
#include <stdint.h>

typedef __attribute__((ext_vector_type(16))) _Float16 v16h;
typedef __attribute__((ext_vector_type(8)))  float    v8f;
typedef __attribute__((ext_vector_type(4)))  unsigned int u32x4;
typedef __attribute__((ext_vector_type(8)))  int          i32x8;
typedef __attribute__((ext_vector_type(4)))  int          i32x4;

#define DD   1024
#define SS   2048
#define NH   16
#define HD   64
#define NROW 8192   // B*S

union AFrag { v16h h; uint32_t u[8]; uint4 q[2]; };

// A-matrix (16x32 f16): lane's packed dwords are {4*hi..4*hi+3} and {8+4*hi..11+4*hi}
__device__ __forceinline__ void load_afrag(AFrag& a, const _Float16* rowptr, int hi) {
    const uint4* p = (const uint4*)rowptr;
    a.q[0] = p[hi];
    a.q[1] = p[2 + hi];
}

__device__ __forceinline__ v8f wmma_f16(const v16h& a, const v16h& b, const v8f& c) {
    return __builtin_amdgcn_wmma_f32_16x16x32_f16(false, a, false, b, (short)0, c, false, false);
}

// ---------------- fp32 -> f16 convert ----------------
__global__ void cvt_f32_f16(const float* __restrict__ in, _Float16* __restrict__ out, int n) {
    int i = blockIdx.x * blockDim.x + threadIdx.x;
    if (i < n) out[i] = (_Float16)in[i];
}

// ---------------- WMMA GEMM: [NROW,1024] @ [1024,1024] ----------------
// MODE 0: f16 -> head layout [B,H,S,HD]      (Q, V projections)
// MODE 1: f32 + bias -> flat [NROW,1024]     (output projection)
// MODE 2: f16 -> transposed head layout [B,H,HD,S]  (K projection, for K^T fragments)
template <int MODE>
__global__ void __launch_bounds__(128)
wmma_gemm(const _Float16* __restrict__ A, const _Float16* __restrict__ Bw,
          _Float16* __restrict__ outh, float* __restrict__ outf,
          const float* __restrict__ bias) {
    const int lane = threadIdx.x & 31;
    const int wid  = (blockIdx.x * blockDim.x + threadIdx.x) >> 5;
    const int rt   = wid >> 4;          // row tile (16 rows), 512 tiles
    const int n0   = (wid & 15) * 64;   // 64-wide col group
    const int m    = lane & 15;
    const int hi   = lane >> 4;

    v8f acc[4] = {v8f{}, v8f{}, v8f{}, v8f{}};

    for (int kb = 0; kb < DD; kb += 32) {
        AFrag a;
        load_afrag(a, A + (size_t)(rt * 16 + m) * DD + kb, hi);
#pragma unroll
        for (int c = 0; c < 4; ++c) {
            v16h bf;
            for (int s = 0; s < 16; ++s) bf[s] = Bw[(size_t)(kb + 8 * hi + s + (s & 8)) * DD + n0 + c * 16 + m];
            acc[c] = wmma_f16(a.h, bf, acc[c]);
        }
    }

#pragma unroll
    for (int c = 0; c < 4; ++c) {
#pragma unroll
        for (int v = 0; v < 8; ++v) {
            const int row = rt * 16 + v + 8 * hi;   // [0,8192)
            const int col = n0 + c * 16 + m;        // [0,1024)
            const int b = row >> 11, s = row & (SS - 1);
            const int h = col >> 6,  d = col & (HD - 1);
            if (MODE == 0) {
                outh[(((size_t)b * NH + h) * SS + s) * HD + d] = (_Float16)acc[c][v];
            } else if (MODE == 2) {
                outh[(((size_t)b * NH + h) * HD + d) * SS + s] = (_Float16)acc[c][v];
            } else {
                outf[(size_t)row * DD + col] = acc[c][v] + bias[col];
            }
        }
    }
}

// ---------------- causal flash attention (one wave per (b,h,q-tile)) ----------------
__global__ void __launch_bounds__(128)
attn_kernel(const _Float16* __restrict__ Qh, const _Float16* __restrict__ Kt,
            const _Float16* __restrict__ Vh, _Float16* __restrict__ ctxh) {
    __shared__ _Float16 plds[4][16 * 32];    // per-wave P staging (C-layout -> A-layout)
    __shared__ _Float16 vlds[4][32 * HD];    // per-wave V tile staged by TDM

    const int lane = threadIdx.x & 31;
    const int wib  = threadIdx.x >> 5;
    const int wid  = blockIdx.x * 4 + wib;
    const int bh   = wid >> 7;      // b*NH + h
    const int q0   = (wid & 127) * 16;
    const int m    = lane & 15;
    const int hi   = lane >> 4;

    const _Float16* Qb  = Qh + (size_t)bh * SS * HD;
    const _Float16* Ktb = Kt + (size_t)bh * HD * SS;   // [HD][S]
    const _Float16* Vb  = Vh + (size_t)bh * SS * HD;

    // Q fragments (A layout) for this 16-row tile, two 32-wide d blocks
    AFrag qf[2];
#pragma unroll
    for (int db = 0; db < 2; ++db)
        load_afrag(qf[db], Qb + (size_t)(q0 + m) * HD + db * 32, hi);

    v8f ctx[4] = {v8f{}, v8f{}, v8f{}, v8f{}};
    float mrun[8], lrun[8];
#pragma unroll
    for (int v = 0; v < 8; ++v) { mrun[v] = -3.0e38f; lrun[v] = 0.0f; }

    _Float16* P  = plds[wib];
    _Float16* Vt = vlds[wib];

    for (int kb = 0; kb < q0 + 16; kb += 32) {
        // all prior LDS reads (P fragments, V fragments) must retire before TDM rewrites Vt
        asm volatile("s_wait_dscnt 0" ::: "memory");

#if __has_builtin(__builtin_amdgcn_tensor_load_to_lds)
        {   // TDM: stage V[kb..kb+31][0..63] (f16) into LDS, D# per ISA 08_async_tensor
            const uint64_t ga = (uint64_t)(uintptr_t)(Vb + (size_t)kb * HD);
            const uint32_t lo = (uint32_t)(uintptr_t)Vt;   // generic LDS addr low bits = byte offset
            u32x4 g0;
            g0[0] = 1u;                                                    // count=1 (valid)
            g0[1] = (uint32_t)__builtin_amdgcn_readfirstlane((int)lo);     // lds_addr
            g0[2] = (uint32_t)__builtin_amdgcn_readfirstlane((int)(uint32_t)ga);
            g0[3] = ((uint32_t)__builtin_amdgcn_readfirstlane(
                        (int)((uint32_t)(ga >> 32) & 0x01FFFFFFu))) | (2u << 30); // type=2
            i32x8 g1;
            g1[0] = 0x00010000;          // workgroup_mask=0, data_size=1 (2 bytes)
            g1[1] = (HD & 0xFFFF) << 16; // tensor_dim0 = 64
            g1[2] = (32 & 0xFFFF) << 16; // tensor_dim1 = 32
            g1[3] = (HD & 0xFFFF) << 16; // tile_dim0 = 64
            g1[4] = 32;                  // tile_dim1 = 32, tile_dim2 = 0
            g1[5] = HD;                  // tensor_dim0_stride = 64
            g1[6] = 0;                   // dim0_stride hi / dim1_stride lo
            g1[7] = 0;
            i32x4 z4 = {0, 0, 0, 0};
            i32x8 z8 = {0, 0, 0, 0, 0, 0, 0, 0};
            __builtin_amdgcn_tensor_load_to_lds(g0, g1, z4, z4, z8, 0);
        }
#endif

        // K^T fragments: strided gather (lane%16 = key col, elements = d rows in table order)
        v16h kf00, kf01, kf10, kf11;
        for (int s = 0; s < 16; ++s) {
            const int d = 8 * hi + s + (s & 8);
            kf00[s] = Ktb[(size_t)(d)      * SS + kb + m];
            kf01[s] = Ktb[(size_t)(32 + d) * SS + kb + m];
            kf10[s] = Ktb[(size_t)(d)      * SS + kb + 16 + m];
            kf11[s] = Ktb[(size_t)(32 + d) * SS + kb + 16 + m];
        }

        v8f s0{}, s1{};
        s0 = wmma_f16(qf[0].h, kf00, s0);
        s0 = wmma_f16(qf[1].h, kf01, s0);
        s1 = wmma_f16(qf[0].h, kf10, s1);
        s1 = wmma_f16(qf[1].h, kf11, s1);

        // scale (1/sqrt(64)) + causal mask, then online softmax
        float f0[8], f1[8], rm[8];
#pragma unroll
        for (int v = 0; v < 8; ++v) {
            const int i  = q0 + v + 8 * hi;
            const int j0 = kb + m, j1 = j0 + 16;
            float a0 = s0[v] * 0.125f; if (j0 > i) a0 = -3.0e38f;
            float a1 = s1[v] * 0.125f; if (j1 > i) a1 = -3.0e38f;
            f0[v] = a0; f1[v] = a1;
            rm[v] = fmaxf(a0, a1);
        }
#pragma unroll
        for (int off = 1; off < 16; off <<= 1)
#pragma unroll
            for (int v = 0; v < 8; ++v) rm[v] = fmaxf(rm[v], __shfl_xor(rm[v], off, 32));

        float rs[8];
#pragma unroll
        for (int v = 0; v < 8; ++v) {
            const float mn    = fmaxf(mrun[v], rm[v]);
            const float alpha = __expf(mrun[v] - mn);
            const float e0 = __expf(f0[v] - mn);
            const float e1 = __expf(f1[v] - mn);
            f0[v] = e0; f1[v] = e1;
            rs[v] = e0 + e1;
            lrun[v] = lrun[v] * alpha;
            mrun[v] = mn;
#pragma unroll
            for (int c = 0; c < 4; ++c) ctx[c][v] *= alpha;
        }
#pragma unroll
        for (int off = 1; off < 16; off <<= 1)
#pragma unroll
            for (int v = 0; v < 8; ++v) rs[v] += __shfl_xor(rs[v], off, 32);
#pragma unroll
        for (int v = 0; v < 8; ++v) lrun[v] += rs[v];

        // P: C-layout -> LDS (row-major 16x32 f16)
#pragma unroll
        for (int v = 0; v < 8; ++v) {
            const int row = v + 8 * hi;
            P[row * 32 + m]      = (_Float16)f0[v];
            P[row * 32 + 16 + m] = (_Float16)f1[v];
        }
        asm volatile("s_wait_dscnt 0" ::: "memory");   // cross-lane LDS visibility

        // P back as A fragment (16 rows x 32 keys): two ds_read_b128
        AFrag pf;
        load_afrag(pf, P + m * 32, hi);

        // V fragments from LDS tile (lane%16 = HD col, elements = key rows in table order)
#if __has_builtin(__builtin_amdgcn_tensor_load_to_lds)
        __builtin_amdgcn_s_wait_tensorcnt(0);
#pragma unroll
        for (int c = 0; c < 4; ++c) {
            v16h vf;
            for (int s = 0; s < 16; ++s) vf[s] = Vt[(size_t)(8 * hi + s + (s & 8)) * HD + c * 16 + m];
            ctx[c] = wmma_f16(pf.h, vf, ctx[c]);
        }
#else
#pragma unroll
        for (int c = 0; c < 4; ++c) {
            v16h vf;
            for (int s = 0; s < 16; ++s) vf[s] = Vb[(size_t)(kb + 8 * hi + s + (s & 8)) * HD + c * 16 + m];
            ctx[c] = wmma_f16(pf.h, vf, ctx[c]);
        }
#endif
    }

    // normalize and store ctx to [B,S,D] f16 for the output projection
    const int b = bh >> 4, h = bh & 15;
#pragma unroll
    for (int v = 0; v < 8; ++v) {
        const float inv = 1.0f / lrun[v];
        const int s = q0 + v + 8 * hi;
#pragma unroll
        for (int c = 0; c < 4; ++c) {
            const int col = h * HD + c * 16 + m;
            ctxh[((size_t)b * SS + s) * DD + col] = (_Float16)(ctx[c][v] * inv);
        }
    }
}

extern "C" void kernel_launch(void* const* d_in, const int* in_sizes, int n_in,
                              void* d_out, int out_size, void* d_ws, size_t ws_size,
                              hipStream_t stream) {
    const float* x  = (const float*)d_in[0];
    const float* Wq = (const float*)d_in[1];
    const float* Wk = (const float*)d_in[2];
    const float* Wv = (const float*)d_in[3];
    const float* Wo = (const float*)d_in[4];
    const float* bo = (const float*)d_in[5];

    char* ws = (char*)d_ws;
    const size_t MB = 1024 * 1024;
    _Float16* xh   = (_Float16*)(ws);              // 16 MB
    _Float16* Wqh  = (_Float16*)(ws + 16 * MB);    //  2 MB each
    _Float16* Wkh  = (_Float16*)(ws + 18 * MB);
    _Float16* Wvh  = (_Float16*)(ws + 20 * MB);
    _Float16* Woh  = (_Float16*)(ws + 22 * MB);
    _Float16* Qh   = (_Float16*)(ws + 24 * MB);    // 16 MB  [B,H,S,HD]
    _Float16* Kth  = (_Float16*)(ws + 40 * MB);    // 16 MB  [B,H,HD,S] (transposed)
    _Float16* Vh   = (_Float16*)(ws + 56 * MB);    // 16 MB  [B,H,S,HD]
    _Float16* ctxh = (_Float16*)(ws + 72 * MB);    // 16 MB  [B,S,D]

    const int NX = NROW * DD;       // 8,388,608
    const int NW = DD * DD;         // 1,048,576
    cvt_f32_f16<<<(NX + 255) / 256, 256, 0, stream>>>(x,  xh,  NX);
    cvt_f32_f16<<<(NW + 255) / 256, 256, 0, stream>>>(Wq, Wqh, NW);
    cvt_f32_f16<<<(NW + 255) / 256, 256, 0, stream>>>(Wk, Wkh, NW);
    cvt_f32_f16<<<(NW + 255) / 256, 256, 0, stream>>>(Wv, Wvh, NW);
    cvt_f32_f16<<<(NW + 255) / 256, 256, 0, stream>>>(Wo, Woh, NW);

    // 8192 waves = 2048 blocks x 4 waves per projection
    wmma_gemm<0><<<2048, 128, 0, stream>>>(xh, Wqh, Qh,  nullptr, nullptr);
    wmma_gemm<2><<<2048, 128, 0, stream>>>(xh, Wkh, Kth, nullptr, nullptr);
    wmma_gemm<0><<<2048, 128, 0, stream>>>(xh, Wvh, Vh,  nullptr, nullptr);

    // B*H*(S/16) = 8192 waves
    attn_kernel<<<2048, 128, 0, stream>>>(Qh, Kth, Vh, ctxh);

    wmma_gemm<1><<<2048, 128, 0, stream>>>(ctxh, Woh, nullptr, (float*)d_out, bo);
}
